// Inter_RNN_35270271435108
// MI455X (gfx1250) — matrix operand order repaired and checked by reading, not compiled, not run on hardware
//
#include <hip/hip_runtime.h>
#include <hip/hip_bf16.h>

typedef unsigned short ushort_t;
typedef unsigned int   uint_t;

typedef __attribute__((ext_vector_type(16))) __bf16 v16bf;
typedef __attribute__((ext_vector_type(8)))  float  v8f;

// Problem constants
#define B_  256
#define T_  200
#define D_  128
#define H_  128
#define G_  896          // 7*H
#define MT_ (B_ * T_)    // 51200 rows for the xg GEMM

union FragBF16 {         // one 16x32 (A) or 32x16 (B) bf16 WMMA operand per lane
    v16bf  v;
    uint4  q[2];
    ushort_t u[16];
};

__device__ __forceinline__ ushort_t f2bf(float f) {
    union { float f; uint_t u; } v; v.f = f;
    uint_t u = v.u;
    return (ushort_t)((u + 0x7FFFu + ((u >> 16) & 1u)) >> 16);  // RNE
}
__device__ __forceinline__ float sigmoidf_(float x) {
    return 1.0f / (1.0f + __expf(-x));
}
__device__ __forceinline__ float softplusf_(float x) {
    return fmaxf(x, 0.0f) + __logf(1.0f + __expf(-fabsf(x)));
}

// ---------------------------------------------------------------------------
// Kernel 1: transpose+convert W_x, W_h ([K=128, N=896] f32 row-major) into
// bf16 [N=896][K=128] so WMMA B-fragments are contiguous 16-byte lane reads along K.
// ---------------------------------------------------------------------------
__global__ __launch_bounds__(256) void convert_weights(
    const float* __restrict__ Wx, const float* __restrict__ Wh,
    ushort_t* __restrict__ WxT, ushort_t* __restrict__ WhT) {
    int idx = blockIdx.x * 256 + threadIdx.x;
    if (idx < G_ * D_) {
        int n = idx / D_, k = idx % D_;
        WxT[idx] = f2bf(Wx[k * G_ + n]);
        WhT[idx] = f2bf(Wh[k * G_ + n]);
    }
}

// ---------------------------------------------------------------------------
// Kernel 2: xg = x @ W_x + b    (M=51200, N=896, K=128), bf16 WMMA, f32 accum.
// One block per 16 M-rows; 8 waves; wave w covers N-tiles w, w+8, ... (<56).
// The 16x128 x-tile is staged once in LDS as bf16 (ISA A-fragment friendly).
// ---------------------------------------------------------------------------
__global__ __launch_bounds__(256) void xg_gemm(
    const float* __restrict__ x, const ushort_t* __restrict__ WxT,
    const float* __restrict__ bias, float* __restrict__ xg) {

    __shared__ ushort_t xa[16 * D_];   // 4 KB

    const int wave = threadIdx.x >> 5;
    const int lane = threadIdx.x & 31;
    const int hs   = lane >> 4;        // lane-half select
    const int l15  = lane & 15;
    const int m0   = blockIdx.x * 16;

    // stage x tile -> bf16 LDS
    for (int i = threadIdx.x; i < 16 * D_; i += 256) {
        xa[i] = f2bf(x[(size_t)(m0 + (i >> 7)) * D_ + (i & 127)]);
    }
    __syncthreads();

    // A fragments for the 4 K-chunks (K=128 = 4 x 32).
    // ISA 16-bit A 16x32 layout: lanes<16 hold K{0..7,16..23}, lanes>=16 +8.
    FragBF16 a[4];
#pragma unroll
    for (int kc = 0; kc < 4; ++kc) {
        const ushort_t* hp = xa + l15 * D_ + kc * 32 + 8 * hs;
        a[kc].q[0] = *(const uint4*)(hp);
        a[kc].q[1] = *(const uint4*)(hp + 16);
    }

    for (int nt = wave; nt < G_ / 16; nt += 8) {
        const int n0 = nt * 16;
        v8f c = {};
#pragma unroll
        for (int kc = 0; kc < 4; ++kc) {
            // B 32x16 layout: lane&15 = column, lanes<16 hold K{0..7,16..23}, lanes>=16 +8
            FragBF16 bf;
            const ushort_t* wp = WxT + (size_t)(n0 + l15) * D_ + kc * 32 + 8 * hs;
            bf.q[0] = *(const uint4*)(wp);
            bf.q[1] = *(const uint4*)(wp + 16);
            c = __builtin_amdgcn_wmma_f32_16x16x32_bf16(
                    false, a[kc].v, false, bf.v, (short)0, c, false, false);
        }
        const float bv = bias[n0 + l15];
#pragma unroll
        for (int r = 0; r < 8; ++r) {
            const int m = m0 + r + 8 * hs;     // C/D: row = r + 8*laneHi
            xg[(size_t)m * G_ + n0 + l15] = c[r] + bv;
        }
    }
}

// ---------------------------------------------------------------------------
// Kernel 3: the CT-LSTM scan. One block per 16 batch rows (16 blocks total).
// W_h (bf16, transposed, 229 KB) lives in LDS for all 200 steps; h ping-pongs
// through a 4 KB bf16 LDS staging tile. Wave w owns H-columns [16w,16w+16);
// all 7 gates for a given (row, col) land in the same lane & register.
// Gather at rep_indices is folded into the scan.
// ---------------------------------------------------------------------------
__global__ __launch_bounds__(256) void ctlstm_scan(
    const float* __restrict__ xg, const float* __restrict__ duration,
    const int* __restrict__ rep, const ushort_t* __restrict__ WhT,
    float* __restrict__ out) {

    extern __shared__ char smem[];
    ushort_t* wh_lds = (ushort_t*)smem;                      // 896*128 bf16 = 229376 B
    ushort_t* h_lds  = (ushort_t*)(smem + G_ * D_ * 2);      // 16*128  bf16 = 4096 B

    const int wave = threadIdx.x >> 5;
    const int lane = threadIdx.x & 31;
    const int hs   = lane >> 4;
    const int l15  = lane & 15;
    const int b0   = blockIdx.x * 16;

    // Pull all of W_h^T into LDS once (uint4 granularity).
    {
        const uint4* src = (const uint4*)WhT;
        uint4*       dst = (uint4*)wh_lds;
        const int nq = G_ * D_ * 2 / 16;
        for (int i = threadIdx.x; i < nq; i += 256) dst[i] = src[i];
    }
    // h0 = 0
    for (int i = threadIdx.x; i < 16 * D_ / 2; i += 256) ((uint_t*)h_lds)[i] = 0u;
    __syncthreads();

    // Per-lane recurrent state: c and cbar for rows (r + 8*hs), col 16*wave+l15.
    v8f cst = {}, cbst = {};
    int repm[8];
#pragma unroll
    for (int r = 0; r < 8; ++r) repm[r] = rep[b0 + r + 8 * hs];

    const int colH = wave * 16 + l15;   // this lane's H column

    for (int t = 0; t < T_; ++t) {
        // A fragments from the bf16 h staging tile.
        FragBF16 a[4];
#pragma unroll
        for (int kc = 0; kc < 4; ++kc) {
            const ushort_t* hp = h_lds + l15 * D_ + kc * 32 + 8 * hs;
            a[kc].q[0] = *(const uint4*)(hp);
            a[kc].q[1] = *(const uint4*)(hp + 16);
        }
        __syncthreads();   // everyone has read h; safe to overwrite below

        // g_j = h @ W_h  for the 7 gates, this wave's 16-column slice of each.
        v8f g[7];
#pragma unroll
        for (int j = 0; j < 7; ++j) {
            const int n0 = j * H_ + wave * 16;
            v8f c = {};
#pragma unroll
            for (int kc = 0; kc < 4; ++kc) {
                FragBF16 bf;
                const ushort_t* wp = wh_lds + (n0 + l15) * D_ + kc * 32 + 8 * hs;
                bf.q[0] = *(const uint4*)(wp);
                bf.q[1] = *(const uint4*)(wp + 16);
                c = __builtin_amdgcn_wmma_f32_16x16x32_bf16(
                        false, a[kc].v, false, bf.v, (short)0, c, false, false);
            }
            g[j] = c;
        }

        // Elementwise CT-LSTM update (all lane-local), + fold in xg_t and the
        // rep_indices gather.
#pragma unroll
        for (int r = 0; r < 8; ++r) {
            const int m   = r + 8 * hs;       // row in tile
            const int bgl = b0 + m;           // global batch index
            const size_t xbase = ((size_t)bgl * T_ + t) * G_ + colH;

            const float gi  = g[0][r] + xg[xbase + 0 * H_];
            const float gf  = g[1][r] + xg[xbase + 1 * H_];
            const float gz  = g[2][r] + xg[xbase + 2 * H_];
            const float go  = g[3][r] + xg[xbase + 3 * H_];
            const float gib = g[4][r] + xg[xbase + 4 * H_];
            const float gfb = g[5][r] + xg[xbase + 5 * H_];
            const float gd  = g[6][r] + xg[xbase + 6 * H_];
            if (t + 1 < T_) __builtin_prefetch(&xg[xbase + G_], 0, 0);

            const float i_  = sigmoidf_(gi);
            const float f_  = sigmoidf_(gf);
            const float o_  = sigmoidf_(go);
            const float ib_ = sigmoidf_(gib);
            const float fb_ = sigmoidf_(gfb);
            const float z_  = tanhf(gz);
            const float dl  = softplusf_(gd);
            const float dt  = duration[(size_t)bgl * T_ + t];

            const float c_ = f_ * cst[r] + i_ * z_;       // cell right after event
            const float cb = fb_ * cbst[r] + ib_ * z_;    // decay target
            const float cn = cb + (c_ - cb) * __expf(-dl * dt);
            const float hn = o_ * tanhf(cn);

            cst[r]  = cn;      // carry is c_new
            cbst[r] = cb;

            h_lds[m * D_ + colH] = f2bf(hn);
            if (repm[r] == t) {
                out[(size_t)bgl * H_ + colH]                = hn;   // h gather
                out[(size_t)B_ * H_ + (size_t)bgl * H_ + colH] = cn; // c gather
            }
        }
        __syncthreads();   // h_lds fully written before next step reads it
    }
}

// ---------------------------------------------------------------------------
extern "C" void kernel_launch(void* const* d_in, const int* in_sizes, int n_in,
                              void* d_out, int out_size, void* d_ws, size_t ws_size,
                              hipStream_t stream) {
    (void)in_sizes; (void)n_in; (void)out_size; (void)ws_size;

    const float* x        = (const float*)d_in[0];   // [B,T,D]
    const float* duration = (const float*)d_in[1];   // [B,T]
    const int*   rep      = (const int*)d_in[2];     // [B]
    const float* Wx       = (const float*)d_in[3];   // [D,7H]
    const float* Wh       = (const float*)d_in[4];   // [H,7H]
    const float* bias     = (const float*)d_in[5];   // [7H]
    float*       out      = (float*)d_out;           // h[1,B,H] ++ c[1,B,H]

    char* ws = (char*)d_ws;
    const size_t XG_BYTES = (size_t)MT_ * G_ * sizeof(float);     // 183.5 MB
    const size_t WT_BYTES = (size_t)G_ * D_ * sizeof(ushort_t);   // 229 KB
    float*    xg  = (float*)ws;
    ushort_t* WxT = (ushort_t*)(ws + XG_BYTES);
    ushort_t* WhT = (ushort_t*)(ws + XG_BYTES + WT_BYTES);

    convert_weights<<<(G_ * D_ + 255) / 256, 256, 0, stream>>>(Wx, Wh, WxT, WhT);

    xg_gemm<<<MT_ / 16, 256, 0, stream>>>(x, WxT, bias, xg);

    const size_t ldsBytes = (size_t)G_ * D_ * 2 + 16 * D_ * 2;    // 233472 B (<320KB WGP LDS)
    (void)hipFuncSetAttribute((const void*)ctlstm_scan,
                              hipFuncAttributeMaxDynamicSharedMemorySize,
                              (int)ldsBytes);
    ctlstm_scan<<<B_ / 16, 256, ldsBytes, stream>>>(xg, duration, rep, WhT, out);
}
